// MultiHeadedAttention_3856880632146
// MI455X (gfx1250) — matrix operand order repaired and checked by reading, not compiled, not run on hardware
//
#include <hip/hip_runtime.h>
#include <stdint.h>

// ---- MHA for MI455X (gfx1250): f16 WMMA + TDM double-buffered GEMMs -------
// B=4, S=2048, D_MODEL=1024, H=16, HEAD=64.

#define S_LEN 2048
#define DM    1024
#define NH    16
#define HD    64
#define MROWS 8192            // B * S

typedef __attribute__((ext_vector_type(16))) _Float16 v16h;
typedef __attribute__((ext_vector_type(8)))  _Float16 h8;
typedef __attribute__((ext_vector_type(8)))  float    v8f;
typedef __attribute__((ext_vector_type(4)))  float    f4;
typedef __attribute__((ext_vector_type(4)))  unsigned int u32x4;
typedef __attribute__((ext_vector_type(4)))  int i32x4;
typedef __attribute__((ext_vector_type(8)))  int i32x8;

union Frag { v16h v; h8 h[2]; };

#define HAVE_TDM __has_builtin(__builtin_amdgcn_tensor_load_to_lds)

static __device__ __forceinline__ v8f wmma_f16(v16h a, v16h b, v8f c) {
    return __builtin_amdgcn_wmma_f32_16x16x32_f16(
        /*neg_a=*/false, a, /*neg_b=*/false, b,
        /*c_mod=*/(short)0, c, /*reuse_a=*/false, /*reuse_b=*/false);
}

static __device__ __forceinline__ void wait_tensorcnt0() {
#if __has_builtin(__builtin_amdgcn_s_wait_tensorcnt)
    __builtin_amdgcn_s_wait_tensorcnt(0);
#else
    asm volatile("s_wait_tensorcnt 0" ::: "memory");
#endif
}

#if HAVE_TDM
// Issue a 2D TDM load: 64 rows x 32 halves, row stride 1024 halves, into LDS.
// D# bitfields per CDNA5 ISA 8.3/8.4 (group0: flags/lds/global addr/type,
// group1: data_size, tensor dims, tile dims, stride).
static __device__ __forceinline__ void tdm_load_tile(
    const _Float16* gsrc, const _Float16* ldst)
{
    unsigned long long ga = (unsigned long long)(uintptr_t)gsrc;
    unsigned lds = (unsigned)(uintptr_t)ldst;   // low 32 bits = LDS offset
    u32x4 g0;
    g0[0] = 1u;                                   // count=1 (valid), user mode
    g0[1] = lds;                                  // lds_addr [63:32]
    g0[2] = (unsigned)ga;                         // global_addr[31:0]
    g0[3] = (unsigned)((ga >> 32) & 0x01FFFFFFu)  // global_addr[56:32]
          | (2u << 30);                           // type = 2 ("image")
    i32x8 g1;
    g1[0] = (int)(1u << 16);                      // data_size=1 -> 2 bytes
    g1[1] = (int)(1024u << 16);                   // tensor_dim0 = 1024
    g1[2] = (int)(0xFFFFu << 16);                 // tensor_dim1 (generous)
    g1[3] = (int)(32u << 16);                     // tile_dim0 = 32 halves
    g1[4] = 64;                                   // tile_dim1 = 64 rows
    g1[5] = 1024;                                 // tensor_dim0_stride = 1024
    g1[6] = 0;
    g1[7] = 0;
    i32x4 z4 = {0, 0, 0, 0};
#if __clang_major__ >= 23
    i32x8 z8 = {0, 0, 0, 0, 0, 0, 0, 0};
    __builtin_amdgcn_tensor_load_to_lds(g0, g1, z4, z4, z8, 0);
#else
    __builtin_amdgcn_tensor_load_to_lds(g0, g1, z4, z4, 0);
#endif
}
#endif

// ---------------------------------------------------------------------------
// Prep kernel: f32 -> f16 elementwise (for X). 8 elements per thread.
// ---------------------------------------------------------------------------
__global__ __launch_bounds__(256) void cvt_f16_kernel(
    const float* __restrict__ in, _Float16* __restrict__ out)
{
    size_t i = ((size_t)blockIdx.x * 256 + threadIdx.x) * 8;
    f4 a = *(const f4*)(in + i);
    f4 b = *(const f4*)(in + i + 4);
    h8 o;
    o[0] = (_Float16)a[0]; o[1] = (_Float16)a[1];
    o[2] = (_Float16)a[2]; o[3] = (_Float16)a[3];
    o[4] = (_Float16)b[0]; o[5] = (_Float16)b[1];
    o[6] = (_Float16)b[2]; o[7] = (_Float16)b[3];
    *(h8*)(out + i) = o;
}

// ---------------------------------------------------------------------------
// Prep kernel: W f32 [k][n] -> Wt f16 [n][k]  (64x64 LDS tile transpose)
// ---------------------------------------------------------------------------
__global__ __launch_bounds__(256) void transpose_w_kernel(
    const float* __restrict__ W, _Float16* __restrict__ Wt)
{
    __shared__ _Float16 tile[64][65];
    const int t  = threadIdx.x;
    const int n0 = blockIdx.x * 64, k0 = blockIdx.y * 64;
    #pragma unroll
    for (int it = 0; it < 4; ++it) {
        int c = it * 256 + t;                 // 1024 float4 chunks
        int row = c >> 4, c4 = c & 15;
        f4 v = *(const f4*)(W + (size_t)(k0 + row) * DM + n0 + c4 * 4);
        #pragma unroll
        for (int e = 0; e < 4; ++e) tile[row][c4 * 4 + e] = (_Float16)v[e];
    }
    __syncthreads();
    #pragma unroll
    for (int it = 0; it < 2; ++it) {
        int c = it * 256 + t;                 // 512 h8 chunks
        int n = c >> 3, ch = c & 7;
        h8 o;
        #pragma unroll
        for (int j = 0; j < 8; ++j) o[j] = tile[ch * 8 + j][n];
        *(h8*)(Wt + (size_t)(n0 + n) * DM + k0 + ch * 8) = o;
    }
}

// ---------------------------------------------------------------------------
// Unified GEMM: C[64x64] = A(f16)[M][1024] @ Bt(f16)[n][k]^T + bias.
// mode 0: f16 out, head-split [B,H,S,64]
// mode 1: f16 out, transposed [B,H,64,S]   (for V)
// mode 2: f32 out, row-major [M][1024]     (final projection)
// TDM double-buffered staging: wave 0 issues tensor_load_to_lds per tile.
// All fragments are loaded before the WMMA chain so a single s_wait_dscnt
// covers the 4 back-to-back WMMAs.
// ---------------------------------------------------------------------------
__global__ __launch_bounds__(128) void gemm_kernel(
    const _Float16* __restrict__ A, const _Float16* __restrict__ Bt,
    const float* __restrict__ bias, void* __restrict__ outp, int mode)
{
    __shared__ _Float16 ldsA[2][64 * 32];
    __shared__ _Float16 ldsB[2][64 * 32];

    const int t    = threadIdx.x;
    const int wave = t >> 5;
    const int lane = t & 31;
    const int hi   = lane >> 4;
    const int ln   = lane & 15;
    const int m0   = blockIdx.x * 64;
    const int n0   = blockIdx.y * 64;
    const int wn   = wave * 16;

    v8f acc[4] = {};

#if HAVE_TDM
    int buf = 0;
    if (wave == 0) {
        tdm_load_tile(A  + (size_t)m0 * DM, &ldsA[0][0]);
        tdm_load_tile(Bt + (size_t)n0 * DM, &ldsB[0][0]);
    }
    for (int kb = 0; kb < DM; kb += 32) {
        wait_tensorcnt0();
        __syncthreads();
        int nxt = buf ^ 1;
        if ((kb + 32 < DM) && wave == 0) {
            tdm_load_tile(A  + (size_t)m0 * DM + kb + 32, &ldsA[nxt][0]);
            tdm_load_tile(Bt + (size_t)n0 * DM + kb + 32, &ldsB[nxt][0]);
        }
        // Load ALL fragments first (10 ds_load_b128), then 4 WMMAs.
        Frag bf, af[4];
        bf.h[0] = *(const h8*)(&ldsB[buf][(wn + ln) * 32 + hi * 8]);
        bf.h[1] = *(const h8*)(&ldsB[buf][(wn + ln) * 32 + hi * 8 + 16]);
        #pragma unroll
        for (int mi = 0; mi < 4; ++mi) {
            af[mi].h[0] = *(const h8*)(&ldsA[buf][(mi * 16 + ln) * 32 + hi * 8]);
            af[mi].h[1] = *(const h8*)(&ldsA[buf][(mi * 16 + ln) * 32 + hi * 8 + 16]);
        }
        #pragma unroll
        for (int mi = 0; mi < 4; ++mi)
            acc[mi] = wmma_f16(af[mi].v, bf.v, acc[mi]);
        buf = nxt;
    }
#else
    for (int kb = 0; kb < DM; kb += 32) {
        #pragma unroll
        for (int it = 0; it < 2; ++it) {
            int c = it * 128 + t;             // 256 h8 chunks per tile
            int row = c >> 2, ch = c & 3;
            *(h8*)(&ldsA[0][row * 32 + ch * 8]) =
                *(const h8*)(A + (size_t)(m0 + row) * DM + kb + ch * 8);
            *(h8*)(&ldsB[0][row * 32 + ch * 8]) =
                *(const h8*)(Bt + (size_t)(n0 + row) * DM + kb + ch * 8);
        }
        __syncthreads();
        Frag bf, af[4];
        bf.h[0] = *(const h8*)(&ldsB[0][(wn + ln) * 32 + hi * 8]);
        bf.h[1] = *(const h8*)(&ldsB[0][(wn + ln) * 32 + hi * 8 + 16]);
        #pragma unroll
        for (int mi = 0; mi < 4; ++mi) {
            af[mi].h[0] = *(const h8*)(&ldsA[0][(mi * 16 + ln) * 32 + hi * 8]);
            af[mi].h[1] = *(const h8*)(&ldsA[0][(mi * 16 + ln) * 32 + hi * 8 + 16]);
        }
        #pragma unroll
        for (int mi = 0; mi < 4; ++mi)
            acc[mi] = wmma_f16(af[mi].v, bf.v, acc[mi]);
        __syncthreads();
    }
#endif

    const int ng = n0 + wn + ln;
    const float bb = bias[ng];
    if (mode == 2) {
        float* out = (float*)outp;
        #pragma unroll
        for (int mi = 0; mi < 4; ++mi)
            #pragma unroll
            for (int r = 0; r < 8; ++r) {
                int m = m0 + mi * 16 + r + hi * 8;
                out[(size_t)m * DM + ng] = acc[mi][r] + bb;
            }
    } else {
        _Float16* out = (_Float16*)outp;
        const int h = ng >> 6, d = ng & 63;
        #pragma unroll
        for (int mi = 0; mi < 4; ++mi)
            #pragma unroll
            for (int r = 0; r < 8; ++r) {
                int m = m0 + mi * 16 + r + hi * 8;
                int b = m >> 11, s = m & 2047;
                size_t idx = mode
                    ? (((size_t)(b * NH + h) * HD + d) * S_LEN + s)
                    : (((size_t)(b * NH + h) * S_LEN + s) * HD + d);
                out[idx] = (_Float16)(acc[mi][r] + bb);
            }
    }
}

// ---------------------------------------------------------------------------
// Flash attention: block = 4 waves x 16 q-rows; keys streamed 32-wide.
// Fragment loads are batched ahead of each WMMA chain.
// ---------------------------------------------------------------------------
__global__ __launch_bounds__(128) void attn_kernel(
    const _Float16* __restrict__ Qh,   // [B,H,S,64] f16
    const _Float16* __restrict__ Kh,   // [B,H,S,64] f16
    const _Float16* __restrict__ Vt,   // [B,H,64,S] f16
    const int* __restrict__ mask,      // [B,S] int32
    _Float16* __restrict__ attended)   // [B,S,H*64] f16
{
    __shared__ _Float16 p_lds[4][16 * 32];   // per-wave P re-layout buffer

    const int t    = threadIdx.x;
    const int wave = t >> 5;
    const int lane = t & 31;
    const int hi   = lane >> 4;
    const int ln   = lane & 15;
    const int h    = blockIdx.y;
    const int b    = blockIdx.z;
    const int q0   = blockIdx.x * 64 + wave * 16;

    const size_t bhS = (size_t)(b * NH + h) * S_LEN;
    const size_t bhD = (size_t)(b * NH + h) * HD;

    Frag qf[2];
    {
        const _Float16* qrow = Qh + (bhS + q0 + ln) * HD;
        #pragma unroll
        for (int dc = 0; dc < 2; ++dc) {
            int base = dc * 32 + hi * 8;
            qf[dc].h[0] = *(const h8*)(qrow + base);
            qf[dc].h[1] = *(const h8*)(qrow + base + 16);
        }
    }

    float m_run[8], l_run[8];
    v8f o_acc[4] = {};
    #pragma unroll
    for (int r = 0; r < 8; ++r) { m_run[r] = -3.0e38f; l_run[r] = 0.f; }

    const float scale = 0.125f;   // 1/sqrt(64)

    for (int kt = 0; kt < S_LEN; kt += 32) {
        if (kt + 32 < S_LEN) {                   // prefetch next K tile
            __builtin_prefetch(Kh + (bhS + kt + 32 + ln) * HD, 0, 0);
        }
        // ---- load all 4 K B-fragments, then 4 chained WMMAs ----
        Frag kf[2][2];
        #pragma unroll
        for (int half = 0; half < 2; ++half) {
            const _Float16* krow = Kh + (bhS + kt + half * 16 + ln) * HD;
            #pragma unroll
            for (int dc = 0; dc < 2; ++dc) {
                int base = dc * 32 + hi * 8;
                kf[half][dc].h[0] = *(const h8*)(krow + base);
                kf[half][dc].h[1] = *(const h8*)(krow + base + 16);
            }
        }
        v8f sc[2] = {};
        #pragma unroll
        for (int half = 0; half < 2; ++half)
            #pragma unroll
            for (int dc = 0; dc < 2; ++dc)
                sc[half] = wmma_f16(qf[dc].v, kf[half][dc].v, sc[half]);

        // ---- scale + key mask ----
        #pragma unroll
        for (int half = 0; half < 2; ++half) {
            bool keep = mask[(size_t)b * S_LEN + kt + half * 16 + ln] != 0;
            #pragma unroll
            for (int r = 0; r < 8; ++r) {
                float v = sc[half][r] * scale;
                sc[half][r] = keep ? v : -3.0e38f;
            }
        }
        // ---- online softmax row stats (16-lane half-wave reductions) ----
        float c_r[8];
        #pragma unroll
        for (int r = 0; r < 8; ++r) {
            float mx = fmaxf(sc[0][r], sc[1][r]);
            #pragma unroll
            for (int off = 8; off >= 1; off >>= 1)
                mx = fmaxf(mx, __shfl_xor(mx, off, 16));
            float m_new = fmaxf(m_run[r], mx);
            float corr  = __expf(m_run[r] - m_new);
            float p0 = __expf(sc[0][r] - m_new);
            float p1 = __expf(sc[1][r] - m_new);
            sc[0][r] = p0; sc[1][r] = p1;
            float rs = p0 + p1;
            #pragma unroll
            for (int off = 8; off >= 1; off >>= 1)
                rs += __shfl_xor(rs, off, 16);
            l_run[r] = l_run[r] * corr + rs;
            m_run[r] = m_new;
            c_r[r]   = corr;
        }
        #pragma unroll
        for (int dt = 0; dt < 4; ++dt)
            #pragma unroll
            for (int r = 0; r < 8; ++r) o_acc[dt][r] *= c_r[r];

        // ---- re-layout P (C layout -> A 16x32 layout) via per-wave LDS ----
        _Float16* pl = &p_lds[wave][0];
        #pragma unroll
        for (int half = 0; half < 2; ++half)
            #pragma unroll
            for (int r = 0; r < 8; ++r)
                pl[(r + hi * 8) * 32 + half * 16 + ln] = (_Float16)sc[half][r];
        asm volatile("s_wait_dscnt 0" ::: "memory");
        Frag pf;
        pf.h[0] = *(const h8*)(pl + ln * 32 + hi * 8);
        pf.h[1] = *(const h8*)(pl + ln * 32 + hi * 8 + 16);

        // ---- load all 4 V B-fragments, then 4 WMMAs ----
        Frag vf[4];
        #pragma unroll
        for (int dt = 0; dt < 4; ++dt) {
            const _Float16* vrow = Vt + (bhD + dt * 16 + ln) * S_LEN
                                      + kt + hi * 8;
            vf[dt].h[0] = *(const h8*)(vrow);
            vf[dt].h[1] = *(const h8*)(vrow + 16);
        }
        #pragma unroll
        for (int dt = 0; dt < 4; ++dt)
            o_acc[dt] = wmma_f16(pf.v, vf[dt].v, o_acc[dt]);
    }

    // ---- finalize: divide by row sum, store merged [B,S,H*64] f16 ----
    float inv[8];
    #pragma unroll
    for (int r = 0; r < 8; ++r) inv[r] = 1.0f / l_run[r];
    #pragma unroll
    for (int dt = 0; dt < 4; ++dt) {
        int dcol = h * HD + dt * 16 + ln;
        #pragma unroll
        for (int r = 0; r < 8; ++r) {
            int sq = q0 + r + hi * 8;
            attended[((size_t)b * S_LEN + sq) * DM + dcol] =
                (_Float16)(o_acc[dt][r] * inv[r]);
        }
    }
}

// ---------------------------------------------------------------------------
extern "C" void kernel_launch(void* const* d_in, const int* in_sizes, int n_in,
                              void* d_out, int out_size, void* d_ws, size_t ws_size,
                              hipStream_t stream) {
    (void)in_sizes; (void)n_in; (void)out_size; (void)ws_size;
    const float* X    = (const float*)d_in[0];
    const int*   mask = (const int*)d_in[1];
    const float* Wq = (const float*)d_in[2]; const float* bq = (const float*)d_in[3];
    const float* Wk = (const float*)d_in[4]; const float* bk = (const float*)d_in[5];
    const float* Wv = (const float*)d_in[6]; const float* bv = (const float*)d_in[7];
    const float* Wo = (const float*)d_in[8]; const float* bo = (const float*)d_in[9];
    float* out = (float*)d_out;

    const size_t XE  = (size_t)MROWS * DM;      // 8388608
    const size_t WE  = (size_t)DM * DM;         // 1048576
    _Float16* Xh  = (_Float16*)d_ws;
    _Float16* Wqt = Xh  + XE;
    _Float16* Wkt = Wqt + WE;
    _Float16* Wvt = Wkt + WE;
    _Float16* Wot = Wvt + WE;
    _Float16* Qh  = Wot + WE;
    _Float16* Kh  = Qh  + XE;
    _Float16* Vt  = Kh  + XE;
    _Float16* att = Vt  + XE;

    // Prep: X -> f16; weights -> transposed f16 [n][k]
    cvt_f16_kernel<<<(unsigned)(XE / (256 * 8)), 256, 0, stream>>>(X, Xh);
    dim3 tgrid(DM / 64, DM / 64);
    transpose_w_kernel<<<tgrid, 256, 0, stream>>>(Wq, Wqt);
    transpose_w_kernel<<<tgrid, 256, 0, stream>>>(Wk, Wkt);
    transpose_w_kernel<<<tgrid, 256, 0, stream>>>(Wv, Wvt);
    transpose_w_kernel<<<tgrid, 256, 0, stream>>>(Wo, Wot);

    dim3 ggrid(MROWS / 64, DM / 64);            // 128 x 16
    gemm_kernel<<<ggrid, 128, 0, stream>>>(Xh, Wqt, bq, (void*)Qh, 0);
    gemm_kernel<<<ggrid, 128, 0, stream>>>(Xh, Wkt, bk, (void*)Kh, 0);
    gemm_kernel<<<ggrid, 128, 0, stream>>>(Xh, Wvt, bv, (void*)Vt, 1);

    dim3 agrid(S_LEN / 64, NH, 4);
    attn_kernel<<<agrid, 128, 0, stream>>>(Qh, Kh, Vt, mask, att);

    gemm_kernel<<<ggrid, 128, 0, stream>>>(att, Wot, bo, (void*)out, 2);
}
